// GQA_12463995093044
// MI455X (gfx1250) — matrix operand order repaired and checked by reading, not compiled, not run on hardware
//
#include <hip/hip_runtime.h>
#include <hip/hip_bf16.h>

// ---------------------------------------------------------------------------
// GQA attention block for MI455X (gfx1250), bf16 WMMA with f32 accumulate.
// Pipeline: cast -> 3x proj GEMM -> RoPE/GQA-mean/permute -> flash attention
//           -> output GEMM.  Matmuls use v_wmma_f32_16x16x32_bf16; LDS tile
//           staging uses GLOBAL_LOAD_ASYNC_TO_LDS_B128 when available.
// ---------------------------------------------------------------------------

typedef __bf16 bf16_t;
typedef __attribute__((ext_vector_type(16))) __bf16 v16bf;
typedef __attribute__((ext_vector_type(8)))  __bf16 v8bf;
typedef __attribute__((ext_vector_type(8)))  float  v8f;
typedef int v4i_vs __attribute__((vector_size(16)));   // matches builtin param type

#define S_LEN    2048
#define DMODEL   1024
#define N_HEADS  16
#define N_GROUPS 4
#define HPG      4
#define HEAD_DIM 64
#define BATCH    2
#define NROWS    (BATCH * S_LEN)   // 4096

#if defined(__gfx1250__) && \
    __has_builtin(__builtin_amdgcn_global_load_async_to_lds_b128) && \
    __has_builtin(__builtin_amdgcn_s_wait_asynccnt)
#define USE_ASYNC 1
#else
#define USE_ASYNC 0
#endif

#define AS1 __attribute__((address_space(1)))
#define AS3 __attribute__((address_space(3)))

// Stage one 16-byte chunk global -> LDS (async on CDNA5, sync fallback).
static __device__ __forceinline__ void stage16(const bf16_t* g, bf16_t* l) {
#if USE_ASYNC
    __builtin_amdgcn_global_load_async_to_lds_b128((AS1 v4i_vs*)(void*)g,
                                                   (AS3 v4i_vs*)l, 0, 0);
#else
    *(v8bf*)l = *(const v8bf*)g;
#endif
}
static __device__ __forceinline__ void stage_wait() {
#if USE_ASYNC
    __builtin_amdgcn_s_wait_asynccnt(0);
#endif
}

// Build a 16-element bf16 fragment from two 16-byte chunks (ds/global b128).
static __device__ __forceinline__ v16bf frag_pair(const bf16_t* p0, const bf16_t* p1) {
    union { v16bf v; v8bf h[2]; } u;
    u.h[0] = *(const v8bf*)p0;
    u.h[1] = *(const v8bf*)p1;
    return u.v;
}

static __device__ __forceinline__ v8f wmma_bf16(v16bf a, v16bf b, v8f c) {
    // (neg_a, A, neg_b, B, c_mod, C, reuse_a, reuse_b)
    return __builtin_amdgcn_wmma_f32_16x16x32_bf16(false, a, false, b, (short)0, c,
                                                   false, false);
}

// ---------------------------------------------------------------------------
// Kernel 1: f32 -> bf16 cast
// ---------------------------------------------------------------------------
__global__ void gqa_cast_bf16(const float* __restrict__ src, bf16_t* __restrict__ dst, int n)
{
    int i = (blockIdx.x * blockDim.x + threadIdx.x) * 4;
    if (i + 3 < n) {
        float4 f = *(const float4*)(src + i);
        dst[i + 0] = (bf16_t)f.x;
        dst[i + 1] = (bf16_t)f.y;
        dst[i + 2] = (bf16_t)f.z;
        dst[i + 3] = (bf16_t)f.w;
    } else {
        for (; i < n; ++i) dst[i] = (bf16_t)src[i];
    }
}

// ---------------------------------------------------------------------------
// Kernel 2/5: Y = X @ W^T, X:[M,K] bf16 row-major, W:[N,K] bf16 row-major.
// 128x128 macro tile, 8 waves (4Mx2N), each wave 32x64 (2x4 16x16 tiles).
// K-step 32, double-buffered LDS filled by async global->LDS copies.
// F32OUT selects f32 vs bf16 stores at compile time.
// ---------------------------------------------------------------------------
template <bool F32OUT>
__global__ __launch_bounds__(256)
void gqa_gemm_bf16(const bf16_t* __restrict__ X, const bf16_t* __restrict__ W,
                   void* __restrict__ Y, int M, int N, int K)
{
    __shared__ __attribute__((aligned(16))) bf16_t Xs[2][128][40];  // +8 pad
    __shared__ __attribute__((aligned(16))) bf16_t Ws[2][128][40];

    const int tid  = threadIdx.x;
    const int lane = tid & 31;
    const int wave = tid >> 5;
    const int wm   = (wave & 3) * 32;   // wave row offset in tile
    const int wn   = (wave >> 2) * 64;  // wave col offset in tile
    const int m0   = blockIdx.y * 128;
    const int n0   = blockIdx.x * 128;

    v8f acc[2][4];
#pragma unroll
    for (int i = 0; i < 2; ++i)
#pragma unroll
        for (int j = 0; j < 4; ++j)
#pragma unroll
            for (int r = 0; r < 8; ++r) acc[i][j][r] = 0.0f;

    // Per-thread chunk mapping: 512 chunks of 8 bf16 per 128x32 tile.
    const int cr0 = (tid)       >> 2, cc0 = ((tid)       & 3) * 8;
    const int cr1 = (tid + 256) >> 2, cc1 = ((tid + 256) & 3) * 8;

    // Prologue: stage k-tile 0 into buffer 0.
    stage16(X + (size_t)(m0 + cr0) * K + cc0, &Xs[0][cr0][cc0]);
    stage16(X + (size_t)(m0 + cr1) * K + cc1, &Xs[0][cr1][cc1]);
    stage16(W + (size_t)(n0 + cr0) * K + cc0, &Ws[0][cr0][cc0]);
    stage16(W + (size_t)(n0 + cr1) * K + cc1, &Ws[0][cr1][cc1]);
    stage_wait();
    __syncthreads();

    const int nk = K >> 5;
    for (int kt = 0; kt < nk; ++kt) {
        const int buf = kt & 1;
        // Kick off the next tile into the other buffer (safe: every wave has
        // passed the barrier that retired its reads of buf^1).
        if (kt + 1 < nk) {
            const int kk = (kt + 1) << 5;
            stage16(X + (size_t)(m0 + cr0) * K + kk + cc0, &Xs[buf ^ 1][cr0][cc0]);
            stage16(X + (size_t)(m0 + cr1) * K + kk + cc1, &Xs[buf ^ 1][cr1][cc1]);
            stage16(W + (size_t)(n0 + cr0) * K + kk + cc0, &Ws[buf ^ 1][cr0][cc0]);
            stage16(W + (size_t)(n0 + cr1) * K + kk + cc1, &Ws[buf ^ 1][cr1][cc1]);
        }
        // A fragments: lane<16 -> row=lane, K 0..7 & 16..23 ; lane>=16 -> K 8..15 & 24..31
        v16bf a[2], b[4];
        {
            const int r  = lane & 15;
            const int kh = (lane < 16) ? 0 : 8;
#pragma unroll
            for (int i = 0; i < 2; ++i) {
                const bf16_t* rp = &Xs[buf][wm + i * 16 + r][kh];
                a[i] = frag_pair(rp, rp + 16);
            }
            // B fragments: lane holds column n = lane&15, K same split as A.
            const int kb = (lane < 16) ? 0 : 8;
#pragma unroll
            for (int j = 0; j < 4; ++j) {
                const bf16_t* rp = &Ws[buf][wn + j * 16 + r][kb];
                b[j] = frag_pair(rp, rp + 16);
            }
        }
#pragma unroll
        for (int i = 0; i < 2; ++i)
#pragma unroll
            for (int j = 0; j < 4; ++j)
                acc[i][j] = wmma_bf16(a[i], b[j], acc[i][j]);
        stage_wait();
        __syncthreads();
    }

    // Epilogue: C/D layout -> VGPR r holds (row r | row 8+r) at col = lane&15.
    const int colbase = n0 + wn + (lane & 15);
    const int rowbase = m0 + wm + ((lane < 16) ? 0 : 8);
#pragma unroll
    for (int i = 0; i < 2; ++i)
#pragma unroll
        for (int r = 0; r < 8; ++r) {
            const size_t rowoff = (size_t)(rowbase + i * 16 + r) * N + colbase;
#pragma unroll
            for (int j = 0; j < 4; ++j) {
                const float v = acc[i][j][r];
                if (F32OUT) ((float*)Y)[rowoff + j * 16]  = v;
                else        ((bf16_t*)Y)[rowoff + j * 16] = (bf16_t)v;
            }
        }
}

// ---------------------------------------------------------------------------
// Kernel 3: RoPE(q,k) + GQA group-mean(k,v) + q head permute.
// grid (S, G, B), 64 threads (one per head dim).
// Outputs: qperm [B][H][S][hd], kg [B][G][S][hd], vgT [B][G][hd][S].
// ---------------------------------------------------------------------------
__global__ __launch_bounds__(64)
void gqa_rope_kernel(const bf16_t* __restrict__ qraw, const bf16_t* __restrict__ kraw,
                     const bf16_t* __restrict__ vraw,
                     bf16_t* __restrict__ qperm, bf16_t* __restrict__ kg,
                     bf16_t* __restrict__ vgT)
{
    const int d = threadIdx.x;          // 0..63
    const int s = blockIdx.x;
    const int g = blockIdx.y;
    const int b = blockIdx.z;
    const int j = d & 31;
    const float inv = __powf(10000.0f, -(float)j / 32.0f);
    const float ang = (float)s * inv;
    const float c  = __cosf(ang);
    const float sn = __sinf(ang);

    const size_t rowoff = (size_t)(b * S_LEN + s) * DMODEL;
    float ksum = 0.0f, vsum = 0.0f;
#pragma unroll
    for (int i = 0; i < HPG; ++i) {
        const int h    = g * HPG + i;
        const int col  = h * HEAD_DIM + d;
        const int colr = h * HEAD_DIM + ((d < 32) ? d + 32 : d - 32);
        const float qv = (float)qraw[rowoff + col];
        const float qr = (float)qraw[rowoff + colr];
        const float kv = (float)kraw[rowoff + col];
        const float kr = (float)kraw[rowoff + colr];
        const float rq = (d < 32) ? -qr : qr;
        const float rk = (d < 32) ? -kr : kr;
        ksum += kv * c + rk * sn;
        vsum += (float)vraw[rowoff + col];
        const int hnew = i * N_GROUPS + g;   // reshape/swapaxes head permute
        qperm[(((size_t)(b * N_HEADS + hnew)) * S_LEN + s) * HEAD_DIM + d] =
            (bf16_t)(qv * c + rq * sn);
    }
    kg[(((size_t)(b * N_GROUPS + g)) * S_LEN + s) * HEAD_DIM + d] = (bf16_t)(ksum * 0.25f);
    vgT[(((size_t)(b * N_GROUPS + g)) * HEAD_DIM + d) * S_LEN + s] = (bf16_t)(vsum * 0.25f);
}

// ---------------------------------------------------------------------------
// Kernel 4: flash attention. grid (S/64, H, B), 128 threads (4 waves).
// Each wave owns 16 q-rows; block processes 64 q-rows against 128-key blocks.
// scores = Q(16x64) @ K^T via 2 WMMAs; online softmax; O += P @ V via WMMAs.
// K/V blocks staged with async global->LDS copies.
// ---------------------------------------------------------------------------
#define KB 128
__global__ __launch_bounds__(128)
void gqa_attn_kernel(const bf16_t* __restrict__ qp, const bf16_t* __restrict__ kg,
                     const bf16_t* __restrict__ vgT, bf16_t* __restrict__ att)
{
    __shared__ __attribute__((aligned(16))) bf16_t Ks[KB][HEAD_DIM + 8];   // [key][d]
    __shared__ __attribute__((aligned(16))) bf16_t Vs[HEAD_DIM][KB + 8];   // [d][key]
    __shared__ __attribute__((aligned(16))) bf16_t Ps[4][16][KB + 8];      // per-wave P

    const int tid  = threadIdx.x;
    const int lane = tid & 31;
    const int wave = tid >> 5;
    const int q0   = blockIdx.x * 64;
    const int h    = blockIdx.y;
    const int b    = blockIdx.z;
    const int g    = h >> 2;  // repeated-kv group for this (permuted) head

    const bf16_t* qbase = qp  + (((size_t)(b * N_HEADS + h)) * S_LEN + q0 + wave * 16) * HEAD_DIM;
    const bf16_t* kbase = kg  + ((size_t)(b * N_GROUPS + g)) * S_LEN * HEAD_DIM;
    const bf16_t* vbase = vgT + ((size_t)(b * N_GROUPS + g)) * HEAD_DIM * S_LEN;

    // Q fragments (A-layout), resident for the whole block.
    v16bf aq[2];
    {
        const int r  = lane & 15;
        const int kh = (lane < 16) ? 0 : 8;
        const bf16_t* rp = qbase + (size_t)r * HEAD_DIM;
        aq[0] = frag_pair(rp + kh,      rp + kh + 16);
        aq[1] = frag_pair(rp + 32 + kh, rp + 32 + kh + 16);
    }

    v8f O[4];
#pragma unroll
    for (int t = 0; t < 4; ++t)
#pragma unroll
        for (int r = 0; r < 8; ++r) O[t][r] = 0.0f;
    float mrow[8], lrow[8];
#pragma unroll
    for (int r = 0; r < 8; ++r) { mrow[r] = -__builtin_inff(); lrow[r] = 0.0f; }

    for (int kb = 0; kb < S_LEN; kb += KB) {
        __syncthreads();   // all waves finished reading previous Ks/Vs
        // Stage K block [KB][hd] and V block (pre-transposed) [hd][KB].
#pragma unroll
        for (int s = 0; s < 8; ++s) {
            int idx = tid + s * 128;          // 1024 chunks of 8 bf16
            int kr = idx >> 3;                // 8 chunks per 64-wide K row
            int kc = (idx & 7) * 8;
            stage16(kbase + (size_t)(kb + kr) * HEAD_DIM + kc, &Ks[kr][kc]);
            int vr = idx >> 4;                // 16 chunks per 128-wide V row
            int vc = (idx & 15) * 8;
            stage16(vbase + (size_t)vr * S_LEN + kb + vc, &Vs[vr][vc]);
        }
        stage_wait();
        __syncthreads();

        // Scores for 8 key sub-tiles of 16.
        v8f sc[8];
        float mb[8];
#pragma unroll
        for (int r = 0; r < 8; ++r) mb[r] = -__builtin_inff();
#pragma unroll
        for (int jj = 0; jj < 8; ++jj) {
            v8f sa;
#pragma unroll
            for (int r = 0; r < 8; ++r) sa[r] = 0.0f;
            const int n  = lane & 15;
            const int dh = (lane < 16) ? 0 : 8;
            const bf16_t* rp0 = &Ks[jj * 16 + n][dh];
            const bf16_t* rp1 = &Ks[jj * 16 + n][32 + dh];
            v16bf b0 = frag_pair(rp0, rp0 + 16);
            v16bf b1 = frag_pair(rp1, rp1 + 16);
            sa = wmma_bf16(aq[0], b0, sa);
            sa = wmma_bf16(aq[1], b1, sa);
#pragma unroll
            for (int r = 0; r < 8; ++r) {
                float v = sa[r] * 0.125f;      // 1/sqrt(64)
                sc[jj][r] = v;
                mb[r] = fmaxf(mb[r], v);
            }
        }
        // Row max across the 16 lanes sharing each row.
#pragma unroll
        for (int r = 0; r < 8; ++r) {
            float v = mb[r];
#pragma unroll
            for (int off = 1; off < 16; off <<= 1)
                v = fmaxf(v, __shfl_xor(v, off, 32));
            mb[r] = v;
        }
        float ls[8];
#pragma unroll
        for (int r = 0; r < 8; ++r) {
            float mn = fmaxf(mrow[r], mb[r]);
            float al = __expf(mrow[r] - mn);
            mrow[r] = mn;
            lrow[r] *= al;
            ls[r] = 0.0f;
#pragma unroll
            for (int t = 0; t < 4; ++t) O[t][r] *= al;
        }
        // exp + write P (bf16) to per-wave LDS for layout conversion C->A.
        const int pcol  = lane & 15;
        const int rbase = (lane < 16) ? 0 : 8;
#pragma unroll
        for (int jj = 0; jj < 8; ++jj) {
#pragma unroll
            for (int r = 0; r < 8; ++r) {
                float p = __expf(sc[jj][r] - mrow[r]);
                ls[r] += p;
                Ps[wave][rbase + r][jj * 16 + pcol] = (bf16_t)p;
            }
        }
#pragma unroll
        for (int r = 0; r < 8; ++r) {
            float v = ls[r];
#pragma unroll
            for (int off = 1; off < 16; off <<= 1)
                v += __shfl_xor(v, off, 32);
            lrow[r] += v;
        }
        // O += P(16xKB) @ V(KBx64): 4 K-steps x 4 output tiles.
#pragma unroll
        for (int f = 0; f < 4; ++f) {
            const int r  = lane & 15;
            const int kh = (lane < 16) ? 0 : 8;
            const bf16_t* rp = &Ps[wave][r][f * 32 + kh];
            v16bf ap = frag_pair(rp, rp + 16);
            const int kk = (lane < 16) ? 0 : 8;
#pragma unroll
            for (int t = 0; t < 4; ++t) {
                const bf16_t* vp = &Vs[t * 16 + r][f * 32 + kk];
                v16bf bv = frag_pair(vp, vp + 16);
                O[t] = wmma_bf16(ap, bv, O[t]);
            }
        }
    }

    // Normalize and store att[b][s][h*64 + d] (bf16).
    const int rowbase = q0 + wave * 16 + ((lane < 16) ? 0 : 8);
    const int colbase = h * HEAD_DIM + (lane & 15);
#pragma unroll
    for (int r = 0; r < 8; ++r) {
        const float inv_l = 1.0f / lrow[r];
        const size_t rowoff = ((size_t)(b * S_LEN) + rowbase + r) * DMODEL + colbase;
#pragma unroll
        for (int t = 0; t < 4; ++t)
            att[rowoff + t * 16] = (bf16_t)(O[t][r] * inv_l);
    }
}

// ---------------------------------------------------------------------------
// Host-side launch
// ---------------------------------------------------------------------------
extern "C" void kernel_launch(void* const* d_in, const int* in_sizes, int n_in,
                              void* d_out, int out_size, void* d_ws, size_t ws_size,
                              hipStream_t stream)
{
    const float* x  = (const float*)d_in[0];
    const float* Wq = (const float*)d_in[1];
    const float* Wk = (const float*)d_in[2];
    const float* Wv = (const float*)d_in[3];
    const float* Wo = (const float*)d_in[4];

    char* w = (char*)d_ws;
    const size_t ND  = (size_t)NROWS * DMODEL;
    const size_t DD  = (size_t)DMODEL * DMODEL;
    const size_t GSZ = (size_t)BATCH * N_GROUPS * S_LEN * HEAD_DIM;

    bf16_t* xb    = (bf16_t*)w; w += ND * 2;
    bf16_t* wqb   = (bf16_t*)w; w += DD * 2;
    bf16_t* wkb   = (bf16_t*)w; w += DD * 2;
    bf16_t* wvb   = (bf16_t*)w; w += DD * 2;
    bf16_t* wob   = (bf16_t*)w; w += DD * 2;
    bf16_t* qraw  = (bf16_t*)w; w += ND * 2;
    bf16_t* kraw  = (bf16_t*)w; w += ND * 2;
    bf16_t* vraw  = (bf16_t*)w; w += ND * 2;
    bf16_t* qperm = (bf16_t*)w; w += ND * 2;
    bf16_t* kgb   = (bf16_t*)w; w += GSZ * 2;
    bf16_t* vgt   = (bf16_t*)w; w += GSZ * 2;
    bf16_t* att   = (bf16_t*)w; w += ND * 2;

    const int nx = (int)ND;
    const int nw = (int)DD;
    gqa_cast_bf16<<<dim3((nx / 4 + 255) / 256), dim3(256), 0, stream>>>(x,  xb,  nx);
    gqa_cast_bf16<<<dim3((nw / 4 + 255) / 256), dim3(256), 0, stream>>>(Wq, wqb, nw);
    gqa_cast_bf16<<<dim3((nw / 4 + 255) / 256), dim3(256), 0, stream>>>(Wk, wkb, nw);
    gqa_cast_bf16<<<dim3((nw / 4 + 255) / 256), dim3(256), 0, stream>>>(Wv, wvb, nw);
    gqa_cast_bf16<<<dim3((nw / 4 + 255) / 256), dim3(256), 0, stream>>>(Wo, wob, nw);

    const dim3 gg(DMODEL / 128, NROWS / 128);  // (8, 32)
    gqa_gemm_bf16<false><<<gg, 256, 0, stream>>>(xb, wqb, qraw, NROWS, DMODEL, DMODEL);
    gqa_gemm_bf16<false><<<gg, 256, 0, stream>>>(xb, wkb, kraw, NROWS, DMODEL, DMODEL);
    gqa_gemm_bf16<false><<<gg, 256, 0, stream>>>(xb, wvb, vraw, NROWS, DMODEL, DMODEL);

    gqa_rope_kernel<<<dim3(S_LEN, N_GROUPS, BATCH), dim3(64), 0, stream>>>(
        qraw, kraw, vraw, qperm, kgb, vgt);

    gqa_attn_kernel<<<dim3(S_LEN / 64, N_HEADS, BATCH), dim3(128), 0, stream>>>(
        qperm, kgb, vgt, att);

    gqa_gemm_bf16<true><<<gg, 256, 0, stream>>>(att, wob, d_out, NROWS, DMODEL, DMODEL);
}
